// GraphAttentionLayer_28767690949412
// MI455X (gfx1250) — matrix operand order repaired and checked by reading, not compiled, not run on hardware
//
#include <hip/hip_runtime.h>
#include <hip/hip_bf16.h>

typedef __attribute__((ext_vector_type(16))) _Float16 v16h;
typedef __attribute__((ext_vector_type(8)))  float    v8f;

#define GAT_N    8192
#define GAT_F    128
#define GAT_ALPHA 0.2f
#define GAT_NEGF (-9.0e15f)

// ---------------------------------------------------------------------------
// Kernel 1: Wh = h @ W (fp32), emitted straight into WMMA-B f16 layout.
//   B tile for K-chunk c (32 rows of Wh), feature slice s (16 cols):
//     lane l (<16: K=0..15, >=16: K=16..31), elem e: K = e + (l>=16)*16, n = l&15
//   Also Wh1[i] = Wh[i,:]·a[:128], Wh2[i] = Wh[i,:]·a[128:].
// ---------------------------------------------------------------------------
__global__ __launch_bounds__(128) void wh_kernel(
    const float* __restrict__ h, const float* __restrict__ W,
    const float* __restrict__ a,
    _Float16* __restrict__ Bws, float* __restrict__ Wh1, float* __restrict__ Wh2)
{
    __shared__ float sh[GAT_F];
    __shared__ float red[GAT_F];
    const int i = blockIdx.x;   // row
    const int f = threadIdx.x;  // feature

    sh[f] = h[(size_t)i * GAT_F + f];
    __syncthreads();

    float acc = 0.0f;
#pragma unroll 8
    for (int k = 0; k < GAT_F; ++k)
        acc = fmaf(sh[k], W[k * GAT_F + f], acc);

    // scatter into WMMA-B layout (f16)
    {
        const int K = i & 31;
        const int c = i >> 5;
        const int l = (f & 15) + (K & 16);       // lane
        const int e = K & 15;                    // element within lane
        const size_t idx = ((size_t)((c << 3) + (f >> 4)) * 32 + l) * 16 + e;
        Bws[idx] = (_Float16)acc;
    }

    // Wh1 reduction
    red[f] = acc * a[f];
    __syncthreads();
    for (int off = 64; off > 0; off >>= 1) {
        if (f < off) red[f] += red[f + off];
        __syncthreads();
    }
    if (f == 0) Wh1[i] = red[0];
    __syncthreads();

    // Wh2 reduction
    red[f] = acc * a[GAT_F + f];
    __syncthreads();
    for (int off = 64; off > 0; off >>= 1) {
        if (f < off) red[f] += red[f + off];
        __syncthreads();
    }
    if (f == 0) Wh2[i] = red[0];
}

// ---------------------------------------------------------------------------
// Kernel 2: per-row softmax stats (max m, sum s of exp(score-m)) over the
// masked score row; streams adj once. Online max/sum + LDS merge.
// ---------------------------------------------------------------------------
__global__ __launch_bounds__(256) void stats_kernel(
    const int* __restrict__ adj, const float* __restrict__ Wh1,
    const float* __restrict__ Wh2, float* __restrict__ rowM, float* __restrict__ rowS)
{
    const int i = blockIdx.x;
    const int t = threadIdx.x;
    const float wh1 = Wh1[i];
    const int* __restrict__ arow = adj + (size_t)i * GAT_N;

    float m = -3.4e38f, s = 0.0f;
    for (int j = t; j < GAT_N; j += 256) {
        float e = wh1 + Wh2[j];
        e = (e > 0.0f) ? e : GAT_ALPHA * e;
        const float sc = (arow[j] > 0) ? e : GAT_NEGF;
        if (sc > m) { s = s * __expf(m - sc); m = sc; }
        s += __expf(sc - m);
    }

    __shared__ float sm[256], ss[256];
    sm[t] = m; ss[t] = s;
    __syncthreads();
    for (int off = 128; off > 0; off >>= 1) {
        if (t < off) {
            const float m2 = sm[t + off], s2 = ss[t + off];
            const float mn = fmaxf(sm[t], m2);
            ss[t] = ss[t] * __expf(sm[t] - mn) + s2 * __expf(m2 - mn);
            sm[t] = mn;
        }
        __syncthreads();
    }
    if (t == 0) { rowM[i] = sm[0]; rowS[i] = ss[0]; }
}

// ---------------------------------------------------------------------------
// Kernel 3: h_prime = softmax(masked scores) @ Wh, then ELU.
// Block: 16 rows x 128 features, 8 waves (wave w -> feature slice w*16).
// Per 32-col K-chunk: 256 threads compute P=exp(score-m) (16x32 f16) directly
// into the WMMA-A register layout in LDS (double buffered), then each wave
// issues one v_wmma_f32_16x16x32_f16 with B from the pre-packed workspace.
// ---------------------------------------------------------------------------
__global__ __launch_bounds__(256) void attn_gemm_kernel(
    const int* __restrict__ adj, const _Float16* __restrict__ Bws,
    const float* __restrict__ Wh1, const float* __restrict__ Wh2,
    const float* __restrict__ rowM, const float* __restrict__ rowS,
    float* __restrict__ out)
{
    __shared__ __align__(32) _Float16 sA[2][32 * 16]; // [buf][lane*16 + elem]
    __shared__ float sWh1[16], sM[16], sS[16];

    const int rowBase = blockIdx.x * 16;
    const int tid  = threadIdx.x;
    const int wave = tid >> 5;     // feature slice 0..7
    const int lane = tid & 31;

    if (tid < 16) {
        sWh1[tid] = Wh1[rowBase + tid];
        sM[tid]   = rowM[rowBase + tid];
        sS[tid]   = rowS[rowBase + tid];
    }
    __syncthreads();

    // P-producer mapping: thread -> (row pr, even k pk); target A slot:
    //   lane pl = pr + (pk&16 ? 16 : 0), elem pe = pk&15
    const int pr = tid >> 4;
    const int pk = (tid & 15) << 1;
    const int pl = pr + ((pk & 16) ? 16 : 0);
    const int pe = pk & 15;
    const float pwh1 = sWh1[pr];
    const float pm   = sM[pr];
    const int* __restrict__ arow = adj + (size_t)(rowBase + pr) * GAT_N + pk;

    v8f acc = {};
    for (int jb = 0, it = 0; jb < GAT_N; jb += 32, ++it) {
        const int buf = it & 1;

        // prefetch adj ~16 chunks ahead (emits global_prefetch_b8)
        __builtin_prefetch(&arow[jb + 512], 0, 1);

        // two probability values for (pr, jb+pk), (pr, jb+pk+1)
        const int a0 = arow[jb];
        const int a1 = arow[jb + 1];
        float e0 = pwh1 + Wh2[jb + pk];
        float e1 = pwh1 + Wh2[jb + pk + 1];
        e0 = (e0 > 0.0f) ? e0 : GAT_ALPHA * e0;
        e1 = (e1 > 0.0f) ? e1 : GAT_ALPHA * e1;
        const float s0 = (a0 > 0) ? e0 : GAT_NEGF;
        const float s1 = (a1 > 0) ? e1 : GAT_NEGF;
        sA[buf][pl * 16 + pe]     = (_Float16)__expf(s0 - pm);
        sA[buf][pl * 16 + pe + 1] = (_Float16)__expf(s1 - pm);

        __syncthreads();

        // A: this wave's 16x32 f16 tile from LDS (already in register layout)
        const v16h A = *(const v16h*)&sA[buf][lane * 16];
        // B: pre-packed Wh chunk tile, coalesced 32B per lane
        const v16h B = *(const v16h*)&Bws[((size_t)(it * 8 + wave) * 32 + lane) * 16];

        acc = __builtin_amdgcn_wmma_f32_16x16x32_f16(
            /*neg_a=*/false, A, /*neg_b=*/false, B,
            /*c_mod=*/(short)0, acc, /*reuse_a=*/false, /*reuse_b=*/false);
    }

    // Epilogue: C layout -> VGPR r holds (M=r, lanes 0-15) and (M=r+8, lanes 16-31)
    const int n    = lane & 15;
    const int mAdd = (lane < 16) ? 0 : 8;
#pragma unroll
    for (int r = 0; r < 8; ++r) {
        const int row = r + mAdd;
        float v = acc[r] / sS[row];
        v = (v > 0.0f) ? v : (__expf(v) - 1.0f);   // ELU (alpha=1)
        out[(size_t)(rowBase + row) * GAT_F + wave * 16 + n] = v;
    }
}

// ---------------------------------------------------------------------------
extern "C" void kernel_launch(void* const* d_in, const int* in_sizes, int n_in,
                              void* d_out, int out_size, void* d_ws, size_t ws_size,
                              hipStream_t stream) {
    (void)in_sizes; (void)n_in; (void)out_size; (void)ws_size;

    const int*   adj = (const int*)d_in[0];
    const float* h   = (const float*)d_in[1];
    const float* W   = (const float*)d_in[2];
    const float* a   = (const float*)d_in[3];
    float* out = (float*)d_out;

    // workspace: Bws (f16, 2MB) | Wh1 | Wh2 | rowM | rowS (32KB each)
    _Float16* Bws = (_Float16*)d_ws;
    float* Wh1  = (float*)((char*)d_ws + ((size_t)GAT_N * GAT_F * sizeof(_Float16)));
    float* Wh2  = Wh1 + GAT_N;
    float* rowM = Wh2 + GAT_N;
    float* rowS = rowM + GAT_N;

    wh_kernel<<<GAT_N, 128, 0, stream>>>(h, W, a, Bws, Wh1, Wh2);
    stats_kernel<<<GAT_N, 256, 0, stream>>>(adj, Wh1, Wh2, rowM, rowS);
    attn_gemm_kernel<<<GAT_N / 16, 256, 0, stream>>>(adj, Bws, Wh1, Wh2, rowM, rowS, out);
}
